// PerformerSelfLayer_30150670418233
// MI455X (gfx1250) — matrix operand order repaired and checked by reading, not compiled, not run on hardware
//
#include <hip/hip_runtime.h>
#include <hip/hip_bf16.h>

// ---------------------------------------------------------------------------
// Performer (FAVOR+) self-attention transformer layer for MI455X (gfx1250).
// All big GEMMs run through V_WMMA_F32_16X16X32_BF16 (bf16 operands, f32 acc)
// with double-buffered LDS staging via the CDNA5 async-copy path
// (GLOBAL_LOAD_ASYNC_TO_LDS_B128 + s_wait_asynccnt), BK=64 per stage.
// ---------------------------------------------------------------------------

typedef __bf16 bf16;
typedef __attribute__((ext_vector_type(4)))  __bf16 bf16x4;
typedef __attribute__((ext_vector_type(8)))  __bf16 bf16x8;
typedef __attribute__((ext_vector_type(16))) __bf16 bf16x16;
typedef __attribute__((ext_vector_type(8)))  float  v8f;
typedef __attribute__((ext_vector_type(4)))  float  f32x4;
typedef __attribute__((ext_vector_type(4)))  int    i32x4;

#define BD      4
#define SEQ     4096
#define DMODEL  1024
#define NHEADS  16
#define DHEAD   64
#define DFF_    4096
#define MFEAT   256

#define NORM_Q   0.35355339059327373f   // 64^{-1/4}
#define NORM2    0.125f                 // 64^{-1/2}
#define RATIO    0.0625f                // 256^{-1/2}
#define EPS_K    1e-4f
#define LN_EPS   1e-5f

#define CAT16(lo, hi) __builtin_shufflevector((lo), (hi), 0,1,2,3,4,5,6,7,8,9,10,11,12,13,14,15)

#if defined(__has_builtin)
#if __has_builtin(__builtin_amdgcn_global_load_async_to_lds_b128)
#define ASYNC_LDS 1
#if __has_builtin(__builtin_amdgcn_s_wait_asynccnt)
#define WAIT_ASYNC() __builtin_amdgcn_s_wait_asynccnt(0)
#else
#define WAIT_ASYNC() asm volatile("s_wait_asynccnt 0" ::: "memory")
#endif
#endif
#endif

// builtin signature wants int4* in the matching address spaces:
//   (global int4* src, lds int4* dst, imm offset, imm cpol)
#define AS1P(p) ((__attribute__((address_space(1))) i32x4*)               \
                 (__attribute__((address_space(1))) void*)(void*)(p))
#define AS3P(p) ((__attribute__((address_space(3))) i32x4*)               \
                 (__attribute__((address_space(3))) void*)(void*)(p))

// epilogue feature flags (compile-time)
#define EP_BIAS  1
#define EP_RELU  2
#define EP_DIV   4
#define EP_OUTF  8
#define EP_OUTB  16

// ---------------------------------------------------------------------------
// Generic batched GEMM:  C[z] = act( A[z] * B[z]^T + bias ) / rowDiv
//   A: [M,K] bf16 row-major (lda), B: [N,K] bf16 row-major (ldb)
//   batch offsets: A += (z/aDiv)*aOuter + (z%aDiv)*aInner,  B += z*bStride,
//                  C += (z/cDiv)*cOuter + (z%cDiv)*cInner
// ---------------------------------------------------------------------------
struct GemmParams {
  const bf16* A; const bf16* B;
  float* Cf; bf16* Cb;
  const float* bias;     // [N]
  const float* rowDiv;   // per-batch, per-row divisor
  int M, N, K;
  int lda, ldb, ldc;
  long long aOuter, aInner; int aDiv;
  long long bStride;
  long long cOuter, cInner; int cDiv;
  long long divStride;
};

template <int BM, int BN, int WM, int WN, int FLAGS>
__global__ __launch_bounds__(256) void gemm_bf16_wmma(const GemmParams p) {
  constexpr int BK = 64;                   // two WMMA K-steps per stage
  constexpr int WAVES_N = BN / WN;
  constexpr int TM = WM / 16;
  constexpr int TN = WN / 16;
  static_assert((BM / WM) * (BN / WN) == 8, "8 waves per block");
  constexpr int LDP = BK + 8;              // padded LDS row (144B stride, 16B aligned)
  constexpr int AU = (BM * BK / 8) / 256;  // A 16B-chunks per thread
  constexpr int BU = (BN * BK / 8) / 256;  // B 16B-chunks per thread
  static_assert(AU >= 1 && BU >= 1, "tile divisible by 256 threads");
  constexpr int CPR = BK / 8;              // 16B chunks per tile row (8)
  constexpr int RSTEP = 256 / CPR;         // rows advanced per chunk step (32)

  constexpr bool HAS_BIAS = (FLAGS & EP_BIAS) != 0;
  constexpr bool HAS_RELU = (FLAGS & EP_RELU) != 0;
  constexpr bool HAS_DIV  = (FLAGS & EP_DIV)  != 0;
  constexpr bool OUT_F    = (FLAGS & EP_OUTF) != 0;
  constexpr bool OUT_B    = (FLAGS & EP_OUTB) != 0;

  __shared__ __align__(16) bf16 As[2][BM][LDP];
  __shared__ __align__(16) bf16 Bs[2][BN][LDP];

  const int tid  = threadIdx.x;
  const int wave = tid >> 5;
  const int lane = tid & 31;
  const int wm = wave / WAVES_N;
  const int wn = wave % WAVES_N;
  const int blockRow = blockIdx.y * BM;
  const int blockCol = blockIdx.x * BN;
  const int z = blockIdx.z;

  const bf16* A = p.A + (long long)(z / p.aDiv) * p.aOuter
                      + (long long)(z % p.aDiv) * p.aInner;
  const bf16* B = p.B + (long long)z * p.bStride;
  const long long cOff = (long long)(z / p.cDiv) * p.cOuter
                       + (long long)(z % p.cDiv) * p.cInner;

  // staging: thread handles chunks (row = tid/8 + u*32, col = (tid&7)*8)
  const int sRow = tid / CPR;
  const int sCol = (tid % CPR) * 8;
  const bf16* aSrc = A + (long long)(blockRow + sRow) * p.lda + sCol;
  const bf16* bSrc = B + (long long)(blockCol + sRow) * p.ldb + sCol;

#ifdef ASYNC_LDS
  auto stage_async = [&](int kk, int buf) {
#pragma unroll
    for (int u = 0; u < AU; ++u)
      __builtin_amdgcn_global_load_async_to_lds_b128(
          AS1P(aSrc + (long long)(u * RSTEP) * p.lda + kk),
          AS3P(&As[buf][sRow + u * RSTEP][sCol]), 0, 0);
#pragma unroll
    for (int u = 0; u < BU; ++u)
      __builtin_amdgcn_global_load_async_to_lds_b128(
          AS1P(bSrc + (long long)(u * RSTEP) * p.ldb + kk),
          AS3P(&Bs[buf][sRow + u * RSTEP][sCol]), 0, 0);
  };
#else
  bf16x8 areg[AU], breg[BU];
  auto stage_load = [&](int kk) {
#pragma unroll
    for (int u = 0; u < AU; ++u)
      areg[u] = *(const bf16x8*)(aSrc + (long long)(u * RSTEP) * p.lda + kk);
#pragma unroll
    for (int u = 0; u < BU; ++u)
      breg[u] = *(const bf16x8*)(bSrc + (long long)(u * RSTEP) * p.ldb + kk);
  };
  auto stage_store = [&](int buf) {
#pragma unroll
    for (int u = 0; u < AU; ++u)
      *(bf16x8*)&As[buf][sRow + u * RSTEP][sCol] = areg[u];
#pragma unroll
    for (int u = 0; u < BU; ++u)
      *(bf16x8*)&Bs[buf][sRow + u * RSTEP][sCol] = breg[u];
  };
#endif

  v8f acc[TM][TN];
#pragma unroll
  for (int i = 0; i < TM; ++i)
#pragma unroll
    for (int j = 0; j < TN; ++j) acc[i][j] = (v8f){};

  const int krow = lane & 15;   // M-row (A) / N-col (B) inside 16x16 subtile
  const int kgrp = lane >> 4;   // which K half this lane owns

  // ---- prologue: tile 0 into buffer 0 -------------------------------------
#ifdef ASYNC_LDS
  stage_async(0, 0);
  WAIT_ASYNC();
#else
  stage_load(0);
  stage_store(0);
#endif
  __syncthreads();

  int buf = 0;
  for (int kk = 0; kk < p.K; kk += BK) {
    const bool hasNext = (kk + BK) < p.K;
    // issue next tile's fetch early so it overlaps the WMMA work below
#ifdef ASYNC_LDS
    if (hasNext) stage_async(kk + BK, buf ^ 1);
#else
    if (hasNext) stage_load(kk + BK);
#endif

#pragma unroll
    for (int ks = 0; ks < BK / 32; ++ks) {
      const int kb = ks * 32;
      // ---- A fragments: lanes 0-15: K 0-7 & 16-23, lanes 16-31: +8 -------
      bf16x16 afrag[TM];
#pragma unroll
      for (int i = 0; i < TM; ++i) {
        int row = wm * WM + i * 16 + krow;
        bf16x8 lo = *(const bf16x8*)&As[buf][row][kb + kgrp * 8];
        bf16x8 hi = *(const bf16x8*)&As[buf][row][kb + 16 + kgrp * 8];
        afrag[i] = CAT16(lo, hi);
      }
      // ---- B fragments: lanes 0-15: K 0-7 & 16-23, lanes 16-31: +8 -------
      bf16x16 bfrag[TN];
#pragma unroll
      for (int j = 0; j < TN; ++j) {
        int col = wn * WN + j * 16 + krow;
        bf16x8 lo = *(const bf16x8*)&Bs[buf][col][kb + kgrp * 8];
        bf16x8 hi = *(const bf16x8*)&Bs[buf][col][kb + 16 + kgrp * 8];
        bfrag[j] = CAT16(lo, hi);
      }
#pragma unroll
      for (int i = 0; i < TM; ++i)
#pragma unroll
        for (int j = 0; j < TN; ++j)
          acc[i][j] = __builtin_amdgcn_wmma_f32_16x16x32_bf16(
              false, afrag[i], false, bfrag[j], (short)0, acc[i][j], false, false);
    }

    if (hasNext) {
#ifdef ASYNC_LDS
      WAIT_ASYNC();               // this wave's async tile writes landed
#else
      stage_store(buf ^ 1);       // s_wait_loadcnt deferred to here
#endif
      __syncthreads();            // publish next buffer / retire this one
      buf ^= 1;
    }
  }

  // ---- epilogue (fully specialized, branch-free) --------------------------
  const int rsel = lane >> 4;     // 0 -> M 0..7, 1 -> M 8..15
  const int ncol = lane & 15;

  float bvals[TN];
#pragma unroll
  for (int j = 0; j < TN; ++j) {
    int col = blockCol + wn * WN + j * 16 + ncol;
    bvals[j] = HAS_BIAS ? p.bias[col] : 0.f;
  }
#pragma unroll
  for (int i = 0; i < TM; ++i) {
#pragma unroll
    for (int e = 0; e < 8; ++e) {
      int row = blockRow + wm * WM + i * 16 + e + 8 * rsel;
      float dscale = 1.f;
      if (HAS_DIV) dscale = 1.f / p.rowDiv[(long long)z * p.divStride + row];
      const long long rbase = cOff + (long long)row * p.ldc;
#pragma unroll
      for (int j = 0; j < TN; ++j) {
        int col = blockCol + wn * WN + j * 16 + ncol;
        float v = acc[i][j][e] + bvals[j];
        if (HAS_RELU) v = fmaxf(v, 0.f);
        if (HAS_DIV) v *= dscale;
        if (OUT_F) p.Cf[rbase + col] = v;
        if (OUT_B) p.Cb[rbase + col] = (bf16)v;
      }
    }
  }
}

// ---------------------------------------------------------------------------
// fp32 -> bf16 conversion (optionally scaled), vectorized x4, grid-stride
// ---------------------------------------------------------------------------
__global__ __launch_bounds__(256) void cvt_kernel(const float* __restrict__ in,
                                                  bf16* __restrict__ out,
                                                  float scale, long long n4) {
  long long i = (long long)blockIdx.x * 256 + threadIdx.x;
  const long long stride = (long long)gridDim.x * 256;
  for (; i < n4; i += stride) {
    f32x4 v = ((const f32x4*)in)[i];
    bf16x4 o;
    o[0] = (bf16)(v.x * scale); o[1] = (bf16)(v.y * scale);
    o[2] = (bf16)(v.z * scale); o[3] = (bf16)(v.w * scale);
    ((bf16x4*)out)[i] = o;
  }
}

__device__ __forceinline__ float waveSum(float v) {
#pragma unroll
  for (int off = 16; off; off >>= 1) v += __shfl_xor(v, off, 32);
  return v;
}
__device__ __forceinline__ float waveMax(float v) {
#pragma unroll
  for (int off = 16; off; off >>= 1) v = fmaxf(v, __shfl_xor(v, off, 32));
  return v;
}

// ---------------------------------------------------------------------------
// FAVOR+ exp:  out = ratio*(exp(xp - diag - stab) + eps), wave per (b,h,n) row
//   stabBH == null  -> query mode (stab = row max)
//   stabBH != null  -> key mode   (stab = per-(b,h) global max)
// ---------------------------------------------------------------------------
__global__ __launch_bounds__(256) void favor_exp_kernel(
    const float* __restrict__ xp, const float* __restrict__ xf,
    const float* __restrict__ stabBH, bf16* __restrict__ outp) {
  const int wave = threadIdx.x >> 5, lane = threadIdx.x & 31;
  const long long r = (long long)blockIdx.x * 8 + wave;   // (b*H+h)*SEQ + n
  const long long n = r % SEQ;
  const long long bh = r / SEQ;
  const int h = (int)(bh % NHEADS);
  const long long b = bh / NHEADS;

  const float* row = xp + r * MFEAT;
  f32x4 x0 = *(const f32x4*)(row + lane * 8);
  f32x4 x1 = *(const f32x4*)(row + lane * 8 + 4);
  float mx = fmaxf(fmaxf(fmaxf(x0.x, x0.y), fmaxf(x0.z, x0.w)),
                   fmaxf(fmaxf(x1.x, x1.y), fmaxf(x1.z, x1.w)));
  mx = waveMax(mx);
  const float stab = stabBH ? stabBH[bh] : mx;

  // diag = 0.5 * norm^2 * sum(x^2) over the head's 64 fp32 values
  const float* qr = xf + (b * SEQ + n) * DMODEL + h * DHEAD + lane * 2;
  float ss = qr[0] * qr[0] + qr[1] * qr[1];
  ss = waveSum(ss);
  const float diag = ss * 0.5f * NORM2;

  float o[8] = {x0.x, x0.y, x0.z, x0.w, x1.x, x1.y, x1.z, x1.w};
  bf16x8 ov;
#pragma unroll
  for (int i = 0; i < 8; ++i)
    ov[i] = (bf16)(RATIO * (__expf(o[i] - diag - stab) + EPS_K));
  *(bf16x8*)(outp + r * MFEAT + lane * 8) = ov;
}

__global__ __launch_bounds__(256) void rowmax_kernel(const float* __restrict__ xp,
                                                     float* __restrict__ rowmax) {
  const int wave = threadIdx.x >> 5, lane = threadIdx.x & 31;
  const long long r = (long long)blockIdx.x * 8 + wave;
  const float* row = xp + r * MFEAT;
  f32x4 x0 = *(const f32x4*)(row + lane * 8);
  f32x4 x1 = *(const f32x4*)(row + lane * 8 + 4);
  float m = fmaxf(fmaxf(fmaxf(x0.x, x0.y), fmaxf(x0.z, x0.w)),
                  fmaxf(fmaxf(x1.x, x1.y), fmaxf(x1.z, x1.w)));
  m = waveMax(m);
  if (lane == 0) rowmax[r] = m;
}

__global__ __launch_bounds__(256) void reduce_max_kernel(const float* __restrict__ rowmax,
                                                         float* __restrict__ stab) {
  __shared__ float sm[256];
  const int bh = blockIdx.x;
  float m = -3.4e38f;
  for (int i = threadIdx.x; i < SEQ; i += 256)
    m = fmaxf(m, rowmax[(long long)bh * SEQ + i]);
  sm[threadIdx.x] = m;
  __syncthreads();
  for (int s = 128; s; s >>= 1) {
    if ((int)threadIdx.x < s) sm[threadIdx.x] = fmaxf(sm[threadIdx.x], sm[threadIdx.x + s]);
    __syncthreads();
  }
  if (threadIdx.x == 0) stab[bh] = sm[0];
}

// k_sum[bh][m] = sum_n kp[bh][n][m]
__global__ __launch_bounds__(256) void ksum_kernel(const bf16* __restrict__ kp,
                                                   float* __restrict__ ksum) {
  const int bh = blockIdx.x;
  const int m = threadIdx.x;
  const bf16* base = kp + (long long)bh * SEQ * MFEAT + m;
  float s = 0.f;
  for (int n = 0; n < SEQ; ++n) s += (float)base[(long long)n * MFEAT];
  ksum[(long long)bh * MFEAT + m] = s;
}

// denom[bh][n] = qp[bh][n][:] . ksum[bh][:]
__global__ __launch_bounds__(256) void denom_kernel(const bf16* __restrict__ qp,
                                                    const float* __restrict__ ksum,
                                                    float* __restrict__ denom) {
  const int wave = threadIdx.x >> 5, lane = threadIdx.x & 31;
  const long long r = (long long)blockIdx.x * 8 + wave;
  const long long bh = r / SEQ;
  bf16x8 qv = *(const bf16x8*)(qp + r * MFEAT + lane * 8);
  const float* ks = ksum + bh * MFEAT + lane * 8;
  float s = 0.f;
#pragma unroll
  for (int i = 0; i < 8; ++i) s += (float)qv[i] * ks[i];
  s = waveSum(s);
  if (lane == 0) denom[r] = s;
}

// ctxT[bh][d][m] = sum_n kp[bh][n][m] * v[b][n][h*64+d]   (VALU reduction)
__global__ __launch_bounds__(256) void ctx_kernel(const bf16* __restrict__ kp,
                                                  const bf16* __restrict__ vb,
                                                  bf16* __restrict__ ctxT) {
  const int bh = blockIdx.x;
  const int h = bh % NHEADS;
  const long long b = bh / NHEADS;
  const int m = threadIdx.x;
  __shared__ float vs[16][DHEAD];
  float acc[DHEAD];
#pragma unroll
  for (int d = 0; d < DHEAD; ++d) acc[d] = 0.f;
  const int lr = threadIdx.x >> 4;
  const int lc = (threadIdx.x & 15) * 4;
  for (int n0 = 0; n0 < SEQ; n0 += 16) {
    bf16x4 vv = *(const bf16x4*)(vb + (b * SEQ + n0 + lr) * DMODEL + h * DHEAD + lc);
    vs[lr][lc + 0] = (float)vv[0];
    vs[lr][lc + 1] = (float)vv[1];
    vs[lr][lc + 2] = (float)vv[2];
    vs[lr][lc + 3] = (float)vv[3];
    __syncthreads();
#pragma unroll 4
    for (int nn = 0; nn < 16; ++nn) {
      float kv = (float)kp[((long long)bh * SEQ + n0 + nn) * MFEAT + m];
#pragma unroll
      for (int d = 0; d < DHEAD; ++d) acc[d] += kv * vs[nn][d];
    }
    __syncthreads();
  }
#pragma unroll
  for (int d = 0; d < DHEAD; ++d)
    ctxT[((long long)bh * DHEAD + d) * MFEAT + m] = (bf16)acc[d];
}

// out = LayerNorm(ra + rb) * g + be ; wave per row of 1024
__global__ __launch_bounds__(256) void ln_kernel(const float* __restrict__ ra,
                                                 const float* __restrict__ rb,
                                                 const float* __restrict__ g,
                                                 const float* __restrict__ be,
                                                 float* __restrict__ outF,
                                                 bf16* __restrict__ outB) {
  const int wave = threadIdx.x >> 5, lane = threadIdx.x & 31;
  const long long r = (long long)blockIdx.x * 8 + wave;
  const float* a = ra + r * DMODEL;
  const float* b = rb + r * DMODEL;
  float vals[32];
  float s = 0.f;
#pragma unroll
  for (int c = 0; c < 8; ++c) {
    int col = c * 128 + lane * 4;
    f32x4 va = *(const f32x4*)(a + col);
    f32x4 vb = *(const f32x4*)(b + col);
    f32x4 x = va + vb;
    vals[c * 4 + 0] = x.x; vals[c * 4 + 1] = x.y;
    vals[c * 4 + 2] = x.z; vals[c * 4 + 3] = x.w;
    s += x.x + x.y + x.z + x.w;
  }
  s = waveSum(s);
  const float mu = s * (1.f / DMODEL);
  float v2 = 0.f;
#pragma unroll
  for (int i = 0; i < 32; ++i) { float d = vals[i] - mu; v2 += d * d; }
  v2 = waveSum(v2);
  const float rstd = rsqrtf(v2 * (1.f / DMODEL) + LN_EPS);
#pragma unroll
  for (int c = 0; c < 8; ++c) {
    int col = c * 128 + lane * 4;
    f32x4 gg = *(const f32x4*)(g + col);
    f32x4 bb = *(const f32x4*)(be + col);
    f32x4 o;
    o.x = (vals[c * 4 + 0] - mu) * rstd * gg.x + bb.x;
    o.y = (vals[c * 4 + 1] - mu) * rstd * gg.y + bb.y;
    o.z = (vals[c * 4 + 2] - mu) * rstd * gg.z + bb.z;
    o.w = (vals[c * 4 + 3] - mu) * rstd * gg.w + bb.w;
    if (outF) *(f32x4*)(outF + r * DMODEL + col) = o;
    if (outB) {
      bf16x4 ob;
      ob[0] = (bf16)o.x; ob[1] = (bf16)o.y; ob[2] = (bf16)o.z; ob[3] = (bf16)o.w;
      *(bf16x4*)(outB + r * DMODEL + col) = ob;
    }
  }
}

// ---------------------------------------------------------------------------
// Host orchestration
// ---------------------------------------------------------------------------
extern "C" void kernel_launch(void* const* d_in, const int* in_sizes, int n_in,
                              void* d_out, int out_size, void* d_ws, size_t ws_size,
                              hipStream_t stream) {
  (void)in_sizes; (void)n_in; (void)out_size; (void)ws_size;
  const float* video = (const float*)d_in[0];
  const float* Wq  = (const float*)d_in[1];
  const float* bq  = (const float*)d_in[2];
  const float* Wk  = (const float*)d_in[3];
  const float* bk  = (const float*)d_in[4];
  const float* Wv  = (const float*)d_in[5];
  const float* bv  = (const float*)d_in[6];
  const float* Wo  = (const float*)d_in[7];
  const float* bo  = (const float*)d_in[8];
  const float* proj = (const float*)d_in[9];
  const float* W1  = (const float*)d_in[10];
  const float* b1  = (const float*)d_in[11];
  const float* W2  = (const float*)d_in[12];
  const float* b2  = (const float*)d_in[13];
  const float* g2  = (const float*)d_in[14];
  const float* be2 = (const float*)d_in[15];
  const float* g3  = (const float*)d_in[16];
  const float* be3 = (const float*)d_in[17];
  float* outp = (float*)d_out;

  const long long ROWS = (long long)BD * SEQ;      // 16384
  const int BH = BD * NHEADS;                      // 64

  char* wsp = (char*)d_ws;
  auto alloc = [&](size_t bytes) -> void* {
    void* p = (void*)wsp;
    wsp += (bytes + 255) & ~(size_t)255;
    return p;
  };

  bf16* vidB  = (bf16*)alloc(ROWS * DMODEL * 2);
  bf16* WqB   = (bf16*)alloc((size_t)DMODEL * DMODEL * 2);
  bf16* WkB   = (bf16*)alloc((size_t)DMODEL * DMODEL * 2);
  bf16* WvB   = (bf16*)alloc((size_t)DMODEL * DMODEL * 2);
  bf16* WoB   = (bf16*)alloc((size_t)DMODEL * DMODEL * 2);
  bf16* W1B   = (bf16*)alloc((size_t)DFF_ * DMODEL * 2);
  bf16* W2B   = (bf16*)alloc((size_t)DMODEL * DFF_ * 2);
  bf16* projB = (bf16*)alloc((size_t)MFEAT * DHEAD * 2);
  float* qF   = (float*)alloc(ROWS * DMODEL * 4);
  float* kF   = (float*)alloc(ROWS * DMODEL * 4);
  bf16* qB    = (bf16*)alloc(ROWS * DMODEL * 2);
  bf16* kB    = (bf16*)alloc(ROWS * DMODEL * 2);
  bf16* vB    = (bf16*)alloc(ROWS * DMODEL * 2);
  float* xpS  = (float*)alloc((size_t)BH * SEQ * MFEAT * 4);   // reused q then k
  bf16* qpB   = (bf16*)alloc((size_t)BH * SEQ * MFEAT * 2);
  bf16* kpB   = (bf16*)alloc((size_t)BH * SEQ * MFEAT * 2);
  float* rmax = (float*)alloc((size_t)BH * SEQ * 4);
  float* stabk = (float*)alloc((size_t)BH * 4);
  float* ksum = (float*)alloc((size_t)BH * MFEAT * 4);
  float* denomBuf = (float*)alloc((size_t)BH * SEQ * 4);
  bf16* ctxT  = (bf16*)alloc((size_t)BH * DHEAD * MFEAT * 2);
  bf16* attnB = (bf16*)alloc(ROWS * DMODEL * 2);
  float* aprj = (float*)alloc(ROWS * DMODEL * 4);
  float* xF   = (float*)alloc(ROWS * DMODEL * 4);
  bf16* xB    = (bf16*)alloc(ROWS * DMODEL * 2);
  bf16* hidB  = (bf16*)alloc(ROWS * DFF_ * 2);
  float* ffF  = (float*)alloc(ROWS * DMODEL * 4);

  auto cvt = [&](const float* in, bf16* out, float scale, long long n) {
    long long n4 = n / 4;
    int grid = (int)((n4 + 255) / 256);
    if (grid > 16384) grid = 16384;
    cvt_kernel<<<grid, 256, 0, stream>>>(in, out, scale, n4);
  };

  // ---- precision conversion ------------------------------------------------
  cvt(video, vidB, 1.f, ROWS * DMODEL);
  cvt(Wq, WqB, 1.f, (long long)DMODEL * DMODEL);
  cvt(Wk, WkB, 1.f, (long long)DMODEL * DMODEL);
  cvt(Wv, WvB, 1.f, (long long)DMODEL * DMODEL);
  cvt(Wo, WoB, 1.f, (long long)DMODEL * DMODEL);
  cvt(W1, W1B, 1.f, (long long)DFF_ * DMODEL);
  cvt(W2, W2B, 1.f, (long long)DMODEL * DFF_);
  cvt(proj, projB, NORM_Q, (long long)MFEAT * DHEAD);   // fold d^{-1/4} into proj

  GemmParams p{};
  p.aDiv = 1; p.cDiv = 1;  // defaults

  // ---- Q, K, V projections -------------------------------------------------
  p.A = vidB; p.lda = DMODEL; p.M = (int)ROWS; p.K = DMODEL; p.N = DMODEL;
  p.ldb = DMODEL; p.ldc = DMODEL;
  p.aOuter = 0; p.aInner = 0; p.bStride = 0; p.cOuter = 0; p.cInner = 0;
  p.divStride = 0; p.rowDiv = nullptr;

  {
    dim3 g(p.N / 128, p.M / 128, 1);
    p.B = WqB; p.bias = bq; p.Cf = qF; p.Cb = qB;
    gemm_bf16_wmma<128, 128, 32, 64, EP_BIAS | EP_OUTF | EP_OUTB><<<g, 256, 0, stream>>>(p);
    p.B = WkB; p.bias = bk; p.Cf = kF; p.Cb = kB;
    gemm_bf16_wmma<128, 128, 32, 64, EP_BIAS | EP_OUTF | EP_OUTB><<<g, 256, 0, stream>>>(p);
    p.B = WvB; p.bias = bv; p.Cf = nullptr; p.Cb = vB;
    gemm_bf16_wmma<128, 128, 32, 64, EP_BIAS | EP_OUTB><<<g, 256, 0, stream>>>(p);
  }

  // ---- FAVOR+ feature GEMMs: xp[bh] = X_head[bh] @ (proj*norm)^T ----------
  GemmParams pf{};
  pf.B = projB; pf.ldb = DHEAD; pf.bStride = 0;
  pf.M = SEQ; pf.N = MFEAT; pf.K = DHEAD; pf.lda = DMODEL; pf.ldc = MFEAT;
  pf.aDiv = NHEADS; pf.aOuter = (long long)SEQ * DMODEL; pf.aInner = DHEAD;
  pf.cDiv = 1; pf.cOuter = (long long)SEQ * MFEAT; pf.cInner = 0;
  pf.bias = nullptr; pf.rowDiv = nullptr; pf.divStride = 0;
  pf.Cf = xpS; pf.Cb = nullptr;

  const int rowBlocks = (int)((long long)BH * SEQ / 8);
  const dim3 gFeat(MFEAT / 128, SEQ / 128, BH);

  // queries
  pf.A = qB;
  gemm_bf16_wmma<128, 128, 32, 64, EP_OUTF><<<gFeat, 256, 0, stream>>>(pf);
  favor_exp_kernel<<<rowBlocks, 256, 0, stream>>>(xpS, qF, nullptr, qpB);

  // keys (global stabilizer over (n, m) per head)
  pf.A = kB;
  gemm_bf16_wmma<128, 128, 32, 64, EP_OUTF><<<gFeat, 256, 0, stream>>>(pf);
  rowmax_kernel<<<rowBlocks, 256, 0, stream>>>(xpS, rmax);
  reduce_max_kernel<<<BH, 256, 0, stream>>>(rmax, stabk);
  favor_exp_kernel<<<rowBlocks, 256, 0, stream>>>(xpS, kF, stabk, kpB);

  // ---- linear attention ----------------------------------------------------
  ksum_kernel<<<BH, 256, 0, stream>>>(kpB, ksum);
  denom_kernel<<<rowBlocks, 256, 0, stream>>>(qpB, ksum, denomBuf);
  ctx_kernel<<<BH, 256, 0, stream>>>(kpB, vB, ctxT);

  // out[bh] = (qp[bh] @ ctxT[bh]^T) / denom ; scattered into [B,N,D] bf16
  GemmParams po{};
  po.A = qpB; po.lda = MFEAT;
  po.aDiv = 1; po.aOuter = (long long)SEQ * MFEAT; po.aInner = 0;
  po.B = ctxT; po.ldb = MFEAT; po.bStride = (long long)DHEAD * MFEAT;
  po.M = SEQ; po.N = DHEAD; po.K = MFEAT;
  po.Cf = nullptr; po.Cb = attnB; po.ldc = DMODEL;
  po.cDiv = NHEADS; po.cOuter = (long long)SEQ * DMODEL; po.cInner = DHEAD;
  po.bias = nullptr;
  po.rowDiv = denomBuf; po.divStride = SEQ;
  {
    dim3 g(po.N / 64, po.M / 128, BH);
    gemm_bf16_wmma<128, 64, 32, 32, EP_DIV | EP_OUTB><<<g, 256, 0, stream>>>(po);
  }

  // ---- output projection + LN1 --------------------------------------------
  p.A = attnB; p.B = WoB; p.bias = bo; p.Cf = aprj; p.Cb = nullptr;
  p.M = (int)ROWS; p.N = DMODEL; p.K = DMODEL;
  p.lda = DMODEL; p.ldb = DMODEL; p.ldc = DMODEL;
  p.rowDiv = nullptr;
  {
    dim3 g(p.N / 128, p.M / 128, 1);
    gemm_bf16_wmma<128, 128, 32, 64, EP_BIAS | EP_OUTF><<<g, 256, 0, stream>>>(p);
  }

  ln_kernel<<<(int)(ROWS / 8), 256, 0, stream>>>(video, aprj, g2, be2, xF, xB);

  // ---- FFN -----------------------------------------------------------------
  p.A = xB; p.B = W1B; p.bias = b1; p.Cf = nullptr; p.Cb = hidB;
  p.M = (int)ROWS; p.N = DFF_; p.K = DMODEL;
  p.lda = DMODEL; p.ldb = DMODEL; p.ldc = DFF_;
  {
    dim3 g(p.N / 128, p.M / 128, 1);
    gemm_bf16_wmma<128, 128, 32, 64, EP_BIAS | EP_RELU | EP_OUTB><<<g, 256, 0, stream>>>(p);
  }

  p.A = hidB; p.B = W2B; p.bias = b2; p.Cf = ffF; p.Cb = nullptr;
  p.M = (int)ROWS; p.N = DMODEL; p.K = DFF_;
  p.lda = DFF_; p.ldb = DFF_; p.ldc = DMODEL;
  {
    dim3 g(p.N / 128, p.M / 128, 1);
    gemm_bf16_wmma<128, 128, 32, 64, EP_BIAS | EP_OUTF><<<g, 256, 0, stream>>>(p);
  }

  // ---- LN2 -> output -------------------------------------------------------
  ln_kernel<<<(int)(ROWS / 8), 256, 0, stream>>>(xF, ffF, g3, be3, outp, nullptr);
}
